// Contrast_49606872269323
// MI455X (gfx1250) — matrix operand order repaired and checked by reading, not compiled, not run on hardware
//
#include <hip/hip_runtime.h>
#include <hip/hip_bf16.h>

typedef __bf16 bf16_t;
typedef __attribute__((ext_vector_type(8)))  __bf16 v8bf;
typedef __attribute__((ext_vector_type(16))) __bf16 v16bf;
typedef __attribute__((ext_vector_type(8)))  float  v8f;

// Build a 16-element bf16 fragment from two 16-byte chunks (ds_load_b128 /
// global_load_b128 pairs). Matches CDNA5 16-bit A-matrix layout where a lane
// holds K = {8*kh..8*kh+7} in v0..v3 and K = {16+8*kh..23+8*kh} in v4..v7.
static __device__ __forceinline__ v16bf frag_split(const bf16_t* lo, const bf16_t* hi) {
    v8bf a = *(const v8bf*)lo;
    v8bf b = *(const v8bf*)hi;
    v16bf r;
#pragma unroll
    for (int i = 0; i < 8; ++i) { r[i] = a[i]; r[i + 8] = b[i]; }
    return r;
}
// B-matrix fragment, p = base + 8*kh: lane holds K = {8*kh..8*kh+7} and {16+8*kh..23+8*kh}.
static __device__ __forceinline__ v16bf frag_contig(const bf16_t* p) {
    return frag_split(p, p + 16);
}

static __device__ __forceinline__ v8f wmma_bf16(v16bf a, v16bf b, v8f c) {
    return __builtin_amdgcn_wmma_f32_16x16x32_bf16(false, a, false, b,
                                                   (short)0, c, false, false);
}

// ---- CDNA5 async global->LDS copy (GLOBAL_LOAD_ASYNC_TO_LDS_B128, GV mode).
// Per lane: LDS[vdst_lane] = MEM[vaddr_lane .. +15]. Tracked by ASYNCcnt.
// The low 32 bits of a generic pointer to __shared__ are the LDS byte offset.
static __device__ __forceinline__ void async_ld_b128(void* lds, const void* g) {
    asm volatile("global_load_async_to_lds_b128 %0, %1, off"
                 :: "v"((unsigned)(unsigned long long)lds), "v"(g) : "memory");
}
static __device__ __forceinline__ void wait_async0() {
    asm volatile("s_wait_asynccnt 0x0" ::: "memory");
}

// ---------------------------------------------------------------- convert ---
__global__ __launch_bounds__(256) void cvt_f32_bf16(const float* __restrict__ in,
                                                    bf16_t* __restrict__ out, int n) {
    int i = blockIdx.x * blockDim.x + threadIdx.x;
    int stride = gridDim.x * blockDim.x;
    for (; i < n; i += stride) out[i] = (bf16_t)in[i];
}

// ------------------------------------------------------------------- GEMM ---
// C[M,N] = act(A[M,K] @ B[K,N] + bias), all bf16 in / bf16 out, f32 accum.
// Block tile 128x128x32, 8 waves, wave tile 32x64 (2x4 of 16x16x32 WMMA).
#define BM 128
#define BN 128
#define BKK 32

__global__ __launch_bounds__(256) void gemm_bf16_kernel(
    const bf16_t* __restrict__ A, int lda,
    const bf16_t* __restrict__ B, int ldb,
    const float* __restrict__ bias,
    bf16_t* __restrict__ C, int ldc,
    int K, int act_elu, int store_trans)
{
    __shared__ bf16_t As[BM][BKK];   // row-major  [m][k]   8 KB
    __shared__ bf16_t Bs[BN][BKK];   // col-major  [n][k]   8 KB

    const int tid  = threadIdx.x;
    const int wave = tid >> 5;
    const int lane = tid & 31;
    const int m0 = blockIdx.y * BM;
    const int n0 = blockIdx.x * BN;
    const int wm = wave >> 1;        // 0..3  (rows)
    const int wn = wave & 1;         // 0..1  (cols)

    v8f acc[2][4] = {};

    const int ar = tid >> 1;             // 0..127
    const int ak = (tid & 1) * 16;
    const int bk = tid >> 3;             // 0..31
    const int bn = (tid & 7) * 16;

    const int r  = lane & 15;
    const int kh = lane >> 4;

    for (int k0 = 0; k0 < K; k0 += BKK) {
        // stage A tile via async DMA to LDS (no VGPR round trip)
        const bf16_t* ag = A + (size_t)(m0 + ar) * lda + k0 + ak;
        async_ld_b128(&As[ar][ak],     ag);
        async_ld_b128(&As[ar][ak + 8], ag + 8);
        // stage B tile transposed -> [n][k] (scatter, must go through VGPRs)
        const bf16_t* bg = B + (size_t)(k0 + bk) * ldb + n0 + bn;
        v8bf b0 = *(const v8bf*)bg;
        v8bf b1 = *(const v8bf*)(bg + 8);
#pragma unroll
        for (int i = 0; i < 8; ++i) {
            Bs[bn + i][bk]     = b0[i];
            Bs[bn + 8 + i][bk] = b1[i];
        }
        if (k0 + BKK < K) {
            __builtin_prefetch(ag + BKK, 0, 1);                               // global_prefetch_b8
            __builtin_prefetch(B + (size_t)(k0 + BKK + bk) * ldb + n0 + bn, 0, 1);
        }
        wait_async0();
        __syncthreads();

        v16bf afr[2];
#pragma unroll
        for (int sm = 0; sm < 2; ++sm) {
            const bf16_t* ap = &As[wm * 32 + sm * 16 + r][0];
            afr[sm] = frag_split(ap + 8 * kh, ap + 16 + 8 * kh);
        }
#pragma unroll
        for (int sn = 0; sn < 4; ++sn) {
            v16bf bfr = frag_contig(&Bs[wn * 64 + sn * 16 + r][8 * kh]);
#pragma unroll
            for (int sm = 0; sm < 2; ++sm)
                acc[sm][sn] = wmma_bf16(afr[sm], bfr, acc[sm][sn]);
        }
        __syncthreads();
    }

    // epilogue: bias, optional ELU, optional transposed store (for V^T)
    const int cl = lane & 15;
    const int rh = kh * 8;
#pragma unroll
    for (int sn = 0; sn < 4; ++sn) {
        const int cg = n0 + wn * 64 + sn * 16 + cl;
        const float bv = bias[cg];
#pragma unroll
        for (int sm = 0; sm < 2; ++sm) {
            const int rbase = m0 + wm * 32 + sm * 16 + rh;
#pragma unroll
            for (int v = 0; v < 8; ++v) {
                float x = acc[sm][sn][v] + bv;
                if (act_elu) x = (x > 0.0f) ? x : (__expf(x) - 1.0f);
                const int rg = rbase + v;
                if (store_trans) C[(size_t)cg * ldc + rg] = (bf16_t)x;
                else             C[(size_t)rg * ldc + cg] = (bf16_t)x;
            }
        }
    }
}

// -------------------------------------------------------- flash attention ---
// Out[q, col_off + d] = softmax(Q K^T / 16) @ V,  D = 512, online softmax.
// 128 queries per block (8 waves x 16 rows), 32-key tiles staged in LDS.
#define AD 512
#define KB 32

__global__ __launch_bounds__(256) void attn_kernel(
    const bf16_t* __restrict__ Q,    // [n][512]
    const bf16_t* __restrict__ Km,   // [n][512]
    const bf16_t* __restrict__ Vt,   // [512][n]  (V transposed)
    float* __restrict__ Out, int ldo, int col_off, int n)
{
    __shared__ bf16_t Ks[KB][AD];    // 32 KB: K tile, rows = key, cols = d
    __shared__ bf16_t Vs[AD][KB];    // 32 KB: V tile, rows = d, cols = key

    const int tid  = threadIdx.x;
    const int wave = tid >> 5;
    const int lane = tid & 31;
    const int q0 = blockIdx.x * 128;

    const int r  = lane & 15;
    const int kh = lane >> 4;

    const bf16_t* qbase = Q + (size_t)(q0 + wave * 16 + r) * AD;

    // per-wave P scratch overlaid on the (already consumed) K tile
    bf16_t* Ps = &Ks[0][0] + wave * (16 * KB);

    float mrow[8], lrow[8];
#pragma unroll
    for (int v = 0; v < 8; ++v) { mrow[v] = -1e30f; lrow[v] = 0.0f; }
    v8f acc[32] = {};   // 16 rows x 512 cols resident (256 VGPRs)

    const float inv_scale = 0.0625f;   // 1/sqrt(512/2)

    const int kr = tid >> 3;           // 0..31 : K-tile staging row
    const int kc = (tid & 7) * 64;
    const int vr = tid * 2;            // 0..510: V-tile staging rows

    for (int j0 = 0; j0 < n; j0 += KB) {
        // ---- stage K and V^T tiles fully asynchronously (ASYNCcnt) ----
        const bf16_t* kg = Km + (size_t)(j0 + kr) * AD + kc;
#pragma unroll
        for (int i = 0; i < 8; ++i)
            async_ld_b128(&Ks[kr][kc + i * 8], kg + i * 8);
#pragma unroll
        for (int d = 0; d < 2; ++d) {
            const bf16_t* vg = Vt + (size_t)(vr + d) * n + j0;
#pragma unroll
            for (int i = 0; i < 4; ++i)
                async_ld_b128(&Vs[vr + d][i * 8], vg + i * 8);
        }
        wait_async0();
        __syncthreads();

        // ---- scores: S[16x32] = Q[16x512] @ K^T tile ----
        v8f s[2] = {};
#pragma unroll 4
        for (int kk = 0; kk < 16; ++kk) {
            v16bf aq = frag_split(qbase + kk * 32 + 8 * kh,
                                  qbase + kk * 32 + 16 + 8 * kh);
#pragma unroll
            for (int sub = 0; sub < 2; ++sub) {
                v16bf bf = frag_contig(&Ks[sub * 16 + r][kk * 32 + 8 * kh]);
                s[sub] = wmma_bf16(aq, bf, s[sub]);
            }
        }

        // ---- online softmax (row = v + 8*kh lives across one 16-lane half) ----
        float p[2][8], mnew[8], alpha[8];
#pragma unroll
        for (int v = 0; v < 8; ++v) {
            float s0 = s[0][v] * inv_scale;
            float s1 = s[1][v] * inv_scale;
            p[0][v] = s0; p[1][v] = s1;
            float mx = fmaxf(s0, s1);
            mx = fmaxf(mx, __shfl_xor(mx, 1));
            mx = fmaxf(mx, __shfl_xor(mx, 2));
            mx = fmaxf(mx, __shfl_xor(mx, 4));
            mx = fmaxf(mx, __shfl_xor(mx, 8));
            mnew[v]  = fmaxf(mrow[v], mx);
            alpha[v] = __expf(mrow[v] - mnew[v]);
            mrow[v]  = mnew[v];
        }
#pragma unroll
        for (int v = 0; v < 8; ++v) {
            float e0 = __expf(p[0][v] - mnew[v]);
            float e1 = __expf(p[1][v] - mnew[v]);
            p[0][v] = e0; p[1][v] = e1;
            float sum = e0 + e1;
            sum += __shfl_xor(sum, 1);
            sum += __shfl_xor(sum, 2);
            sum += __shfl_xor(sum, 4);
            sum += __shfl_xor(sum, 8);
            lrow[v] = lrow[v] * alpha[v] + sum;
        }
#pragma unroll
        for (int t = 0; t < 32; ++t)
#pragma unroll
            for (int v = 0; v < 8; ++v)
                acc[t][v] = acc[t][v] * alpha[v];

        __syncthreads();   // all waves done reading Ks before P overlay write

        // ---- C-layout -> A-fragment relayout of P via per-wave LDS ----
#pragma unroll
        for (int sub = 0; sub < 2; ++sub) {
            const int col = sub * 16 + (lane & 15);
#pragma unroll
            for (int v = 0; v < 8; ++v)
                Ps[(v + kh * 8) * KB + col] = (bf16_t)p[sub][v];
        }
        v16bf pf = frag_split(Ps + r * KB + 8 * kh, Ps + r * KB + 16 + 8 * kh);

        // ---- O += P[16x32] @ V[32x512] ----
#pragma unroll 8
        for (int t = 0; t < 32; ++t) {
            v16bf bv = frag_contig(&Vs[t * 16 + r][8 * kh]);
            acc[t] = wmma_bf16(pf, bv, acc[t]);
        }
        __syncthreads();   // before next tile overwrites Ks/Vs/Ps
    }

    // ---- normalize + store f32 output ----
#pragma unroll
    for (int v = 0; v < 8; ++v) {
        const float invl = 1.0f / lrow[v];
        const int rg = q0 + wave * 16 + v + kh * 8;
#pragma unroll
        for (int t = 0; t < 32; ++t) {
            const int cg = col_off + t * 16 + (lane & 15);
            Out[(size_t)rg * ldo + cg] = acc[t][v] * invl;
        }
    }
}

// ---------------------------------------------------------------- launch ----
extern "C" void kernel_launch(void* const* d_in, const int* in_sizes, int n_in,
                              void* d_out, int out_size, void* d_ws, size_t ws_size,
                              hipStream_t stream) {
    const int N = 8192, HID = 1024, OUTD = 512;

    const float* za = (const float*)d_in[0];
    const float* zb = (const float*)d_in[1];
    const float* W1 = (const float*)d_in[2];
    const float* b1 = (const float*)d_in[3];
    const float* W2 = (const float*)d_in[4];
    const float* b2 = (const float*)d_in[5];
    const float* Wq = (const float*)d_in[6];
    const float* bq = (const float*)d_in[7];
    const float* Wk = (const float*)d_in[8];
    const float* bk = (const float*)d_in[9];
    const float* Wv = (const float*)d_in[10];
    const float* bv = (const float*)d_in[11];
    float* out = (float*)d_out;

    char* ws = (char*)d_ws;
    const size_t MB = 1u << 20;
    bf16_t* zaBf = (bf16_t*)(ws + 0 * MB);     // 16 MB, reused as Qa
    bf16_t* zbBf = (bf16_t*)(ws + 16 * MB);    // 16 MB, reused as Qb
    bf16_t* Ha   = (bf16_t*)(ws + 32 * MB);    // 16 MB, reused as Ka/Kb
    bf16_t* Hb   = (bf16_t*)(ws + 48 * MB);    // 16 MB, reused as VaT/VbT
    bf16_t* Pa   = (bf16_t*)(ws + 64 * MB);    // 8 MB
    bf16_t* Pb   = (bf16_t*)(ws + 72 * MB);    // 8 MB
    bf16_t* W1b  = (bf16_t*)(ws + 80 * MB);    // 2 MB
    bf16_t* W2b  = (bf16_t*)(ws + 82 * MB);    // 1 MB
    bf16_t* Wqb  = (bf16_t*)(ws + 83 * MB);    // 0.5 MB
    bf16_t* Wkb  = (bf16_t*)(ws + 83 * MB + 512 * 1024);
    bf16_t* Wvb  = (bf16_t*)(ws + 84 * MB);
    bf16_t* Qa = zaBf, *Qb = zbBf;
    bf16_t* Ka = Ha;
    bf16_t* Kb  = (bf16_t*)(ws + 40 * MB);
    bf16_t* VaT = Hb;
    bf16_t* VbT = (bf16_t*)(ws + 56 * MB);

    // 1) f32 -> bf16 conversions
    cvt_f32_bf16<<<1024, 256, 0, stream>>>(za, zaBf, N * HID);
    cvt_f32_bf16<<<1024, 256, 0, stream>>>(zb, zbBf, N * HID);
    cvt_f32_bf16<<<512,  256, 0, stream>>>(W1, W1b, HID * HID);
    cvt_f32_bf16<<<512,  256, 0, stream>>>(W2, W2b, HID * OUTD);
    cvt_f32_bf16<<<256,  256, 0, stream>>>(Wq, Wqb, OUTD * OUTD);
    cvt_f32_bf16<<<256,  256, 0, stream>>>(Wk, Wkb, OUTD * OUTD);
    cvt_f32_bf16<<<256,  256, 0, stream>>>(Wv, Wvb, OUTD * OUTD);

    // 2) H = elu(z @ W1 + b1)
    gemm_bf16_kernel<<<dim3(HID / BN, N / BM), 256, 0, stream>>>(
        zaBf, HID, W1b, HID, b1, Ha, HID, HID, 1, 0);
    gemm_bf16_kernel<<<dim3(HID / BN, N / BM), 256, 0, stream>>>(
        zbBf, HID, W1b, HID, b1, Hb, HID, HID, 1, 0);

    // 3) P = H @ W2 + b2
    gemm_bf16_kernel<<<dim3(OUTD / BN, N / BM), 256, 0, stream>>>(
        Ha, HID, W2b, OUTD, b2, Pa, OUTD, HID, 0, 0);
    gemm_bf16_kernel<<<dim3(OUTD / BN, N / BM), 256, 0, stream>>>(
        Hb, HID, W2b, OUTD, b2, Pb, OUTD, HID, 0, 0);

    // 4) Q/K/V projections (V stored transposed [512][8192])
    gemm_bf16_kernel<<<dim3(OUTD / BN, N / BM), 256, 0, stream>>>(
        Pa, OUTD, Wqb, OUTD, bq, Qa, OUTD, OUTD, 0, 0);
    gemm_bf16_kernel<<<dim3(OUTD / BN, N / BM), 256, 0, stream>>>(
        Pa, OUTD, Wkb, OUTD, bk, Ka, OUTD, OUTD, 0, 0);
    gemm_bf16_kernel<<<dim3(OUTD / BN, N / BM), 256, 0, stream>>>(
        Pa, OUTD, Wvb, OUTD, bv, VaT, N, OUTD, 0, 1);
    gemm_bf16_kernel<<<dim3(OUTD / BN, N / BM), 256, 0, stream>>>(
        Pb, OUTD, Wqb, OUTD, bq, Qb, OUTD, OUTD, 0, 0);
    gemm_bf16_kernel<<<dim3(OUTD / BN, N / BM), 256, 0, stream>>>(
        Pb, OUTD, Wkb, OUTD, bk, Kb, OUTD, OUTD, 0, 0);
    gemm_bf16_kernel<<<dim3(OUTD / BN, N / BM), 256, 0, stream>>>(
        Pb, OUTD, Wvb, OUTD, bv, VbT, N, OUTD, 0, 1);

    // 5) cross-attention both directions -> out[:, 0:512] and out[:, 512:1024]
    attn_kernel<<<N / 128, 256, 0, stream>>>(Qa, Kb, VbT, out, 2 * OUTD, 0,    N);
    attn_kernel<<<N / 128, 256, 0, stream>>>(Qb, Ka, VaT, out, 2 * OUTD, OUTD, N);
}
